// Model_52407190946026
// MI455X (gfx1250) — matrix operand order repaired and checked by reading, not compiled, not run on hardware
//
#include <hip/hip_runtime.h>
#include <hip/hip_bf16.h>

typedef __attribute__((ext_vector_type(8)))  float  v8f;
typedef __attribute__((ext_vector_type(8)))  __bf16 v8bf;
typedef __attribute__((ext_vector_type(16))) __bf16 v16bf;

#define SK 40            // generic-GEMM LDS row stride (shorts), keeps 16B align
#define HS 264           // h-cache row stride (shorts): 256 + 8 pad, 528B (16B-aligned)
#define SLOPE 0.01f

__device__ __forceinline__ unsigned short f2bf(float f) {
  unsigned int u = __float_as_uint(f);
  u += 0x7FFFu + ((u >> 16) & 1u);       // round-to-nearest-even
  return (unsigned short)(u >> 16);
}
__device__ __forceinline__ unsigned int pack2(float a, float b) {
  return (unsigned int)f2bf(a) | ((unsigned int)f2bf(b) << 16);
}
__device__ __forceinline__ v16bf frag16(const unsigned short* p0, const unsigned short* p1) {
  v8bf lo = *reinterpret_cast<const v8bf*>(p0);   // b128 load
  v8bf hi = *reinterpret_cast<const v8bf*>(p1);   // b128 load
  return __builtin_shufflevector(lo, hi, 0,1,2,3,4,5,6,7,8,9,10,11,12,13,14,15);
}

// ---------------------------------------------------------------------------
// Generic tiled GEMM: C[M,N] = act(A[M,K] @ B[K,N] + bias)   (minor cost path)
// fp32 in/out, bf16 WMMA compute. 128x128 tile, BK=32, 256 threads = 8 waves.
// ---------------------------------------------------------------------------
template<int ACT, int BIAS>
__global__ __launch_bounds__(256, 1)
void gemm_wmma(const float* __restrict__ A, int lda,
               const float* __restrict__ B, int ldb,
               const float* __restrict__ bias,
               float* __restrict__ C, int ldc,
               int M, int N, int K)
{
  __shared__ unsigned short As[128 * SK];   // [row][k]
  __shared__ unsigned short Bs[128 * SK];   // transposed: [n][k]

  const int tid  = threadIdx.x;
  const int lane = tid & 31;
  const int wave = tid >> 5;
  const int wm   = wave & 3;
  const int wn   = wave >> 2;
  const int m0   = blockIdx.y * 128;
  const int n0   = blockIdx.x * 128;

  const int arow = tid >> 3;
  const int acol = (tid & 7) * 4;
  const int brow = tid >> 5;
  const int bcol = (tid & 31) * 4;
  const int lhalf = lane >> 4;
  const int l16   = lane & 15;

  v8f acc[2][4] = {};

  for (int k0 = 0; k0 < K; k0 += 32) {
    const int krem = K - k0;
#pragma unroll
    for (int it = 0; it < 4; ++it) {
      const int r  = it * 32 + arow;
      const int gr = m0 + r;
      float4 v = make_float4(0.f, 0.f, 0.f, 0.f);
      if (gr < M) {
        const float* p = A + (size_t)gr * lda + (k0 + acol);
        if (acol + 4 <= krem) {
          v = *reinterpret_cast<const float4*>(p);
        } else {
          if (acol + 0 < krem) v.x = p[0];
          if (acol + 1 < krem) v.y = p[1];
          if (acol + 2 < krem) v.z = p[2];
          if (acol + 3 < krem) v.w = p[3];
        }
      }
      *reinterpret_cast<uint2*>(&As[r * SK + acol]) =
          make_uint2(pack2(v.x, v.y), pack2(v.z, v.w));
    }
#pragma unroll
    for (int it = 0; it < 4; ++it) {
      const int r  = it * 8 + brow;
      const int gk = k0 + r;
      const int gn = n0 + bcol;
      float4 v = make_float4(0.f, 0.f, 0.f, 0.f);
      if (gk < K) {
        const float* p = B + (size_t)gk * ldb + gn;
        if (gn + 4 <= N) {
          v = *reinterpret_cast<const float4*>(p);
        } else {
          if (gn + 0 < N) v.x = p[0];
          if (gn + 1 < N) v.y = p[1];
          if (gn + 2 < N) v.z = p[2];
          if (gn + 3 < N) v.w = p[3];
        }
      }
      Bs[(bcol + 0) * SK + r] = f2bf(v.x);
      Bs[(bcol + 1) * SK + r] = f2bf(v.y);
      Bs[(bcol + 2) * SK + r] = f2bf(v.z);
      Bs[(bcol + 3) * SK + r] = f2bf(v.w);
    }
    __syncthreads();

    v16bf afrag[2], bfrag[4];
#pragma unroll
    for (int fm = 0; fm < 2; ++fm) {
      const unsigned short* base = &As[(wm * 32 + fm * 16 + l16) * SK];
      afrag[fm] = frag16(base + lhalf * 8, base + lhalf * 8 + 16);
    }
#pragma unroll
    for (int fn = 0; fn < 4; ++fn) {
      const unsigned short* base = &Bs[(wn * 64 + fn * 16 + l16) * SK + lhalf * 8];
      bfrag[fn] = frag16(base, base + 16);
    }
#pragma unroll
    for (int fm = 0; fm < 2; ++fm)
#pragma unroll
      for (int fn = 0; fn < 4; ++fn)
        acc[fm][fn] = __builtin_amdgcn_wmma_f32_16x16x32_bf16(
            false, afrag[fm], false, bfrag[fn], (short)0, acc[fm][fn], false, false);
    __syncthreads();
  }

#pragma unroll
  for (int fm = 0; fm < 2; ++fm) {
    const int rbase = m0 + wm * 32 + fm * 16 + lhalf * 8;
#pragma unroll
    for (int fn = 0; fn < 4; ++fn) {
      const int col = n0 + wn * 64 + fn * 16 + l16;
      if (col < N) {
        const float bv = BIAS ? bias[col] : 0.f;
#pragma unroll
        for (int v = 0; v < 8; ++v) {
          const int row = rbase + v;
          if (row < M) {
            float x = acc[fm][fn][v] + bv;
            if (ACT) x = (x >= 0.f) ? x : SLOPE * x;
            C[(size_t)row * ldc + col] = x;
          }
        }
      }
    }
  }
}

// ---------------------------------------------------------------------------
// Repack W[256, 65536] fp32 (+ bias bW[65536]) into chunk-blocked transposed
// bf16:  Wt[c*8192 + o*32 + k'] = bf16(B'(kk = c*32+k', o)), c in [0, 2056)
//   c < 2048 : B'(kk,o) = W[(kk>>8)*65536 + (kk&255)*256 + o]
//   c >= 2048: B'(kk,o) = bW[(kk-65536)*256 + o]
// LDS-tiled so both global read and write are fully coalesced.
// ---------------------------------------------------------------------------
__global__ __launch_bounds__(256, 1)
void repack_w(const float* __restrict__ W, const float* __restrict__ bW,
              unsigned short* __restrict__ Wt)
{
  __shared__ unsigned short T[32 * HS];   // [k'][o]
  const int c = blockIdx.x;
  const int t = threadIdx.x;

  const int r  = t >> 3;          // k' row 0..31
  const int c4 = (t & 7) * 4;     // o sub-offset
  const int kk = c * 32 + r;
  const float* src = (kk < 65536)
      ? (W + (size_t)(kk >> 8) * 65536 + (size_t)(kk & 255) * 256)
      : (bW + (size_t)(kk - 65536) * 256);
#pragma unroll
  for (int it = 0; it < 8; ++it) {
    const int o = c4 + it * 32;
    const float4 v = *reinterpret_cast<const float4*>(src + o);
    *reinterpret_cast<uint2*>(&T[r * HS + o]) =
        make_uint2(pack2(v.x, v.y), pack2(v.z, v.w));
  }
  __syncthreads();

  unsigned short* dst = Wt + (size_t)c * 8192 + (size_t)t * 32;   // o = t
#pragma unroll
  for (int j = 0; j < 4; ++j) {
    unsigned int w0 = (unsigned int)T[(j*8+0)*HS + t] | ((unsigned int)T[(j*8+1)*HS + t] << 16);
    unsigned int w1 = (unsigned int)T[(j*8+2)*HS + t] | ((unsigned int)T[(j*8+3)*HS + t] << 16);
    unsigned int w2 = (unsigned int)T[(j*8+4)*HS + t] | ((unsigned int)T[(j*8+5)*HS + t] << 16);
    unsigned int w3 = (unsigned int)T[(j*8+6)*HS + t] | ((unsigned int)T[(j*8+7)*HS + t] << 16);
    *reinterpret_cast<uint4*>(dst + j * 8) = make_uint4(w0, w1, w2, w3);
  }
}

// ---------------------------------------------------------------------------
// Fused NNConv message pass (dominant kernel).
//   msg[e,o] = sum_{kz,i} z[e,kz]*h[src_e,i]*W[kz, i*256+o] + sum_i h[src_e,i]*bW[i*256+o]
// A-side: 128 gathered h rows cached in LDS (bf16) once; A-frag = ds_load_b128
//         pair scaled per-lane by bf16(z[e,kz])  (rank-1 trick -> no restaging).
// B-side: direct coalesced global_load_b128 pairs from the repacked Wt (bf16).
// NO barriers in the 2056-chunk main loop.
// Epilogue: atomic scatter-add into agg[dst_e].
// ---------------------------------------------------------------------------
__global__ __launch_bounds__(256, 1)
void nnconv_msg(const float* __restrict__ hsrc,          // [*, 256] fp32
                const float* __restrict__ z,             // [E, 256] fp32
                const unsigned short* __restrict__ Wt,   // [2056*8192] bf16 repacked
                const int* __restrict__ src,
                const int* __restrict__ dst,
                float* __restrict__ agg,                 // [S, 256] atomic accum
                int E)
{
  __shared__ unsigned short Hc[128 * HS];   // gathered h rows, bf16

  const int tid  = threadIdx.x;
  const int lane = tid & 31;
  const int wave = tid >> 5;
  const int wm   = wave & 3;       // 4 waves along M (edges)
  const int wn   = wave >> 2;      // 2 waves along N (out cols)
  const int m0   = blockIdx.y * 128;
  const int n0   = blockIdx.x * 128;
  const int lhalf = lane >> 4;
  const int l16   = lane & 15;

  // ---- prologue: gather + convert 128 h rows into LDS (2 threads / row) ----
  {
    const int r  = tid >> 1;
    const int cb = (tid & 1) * 128;
    const int ge = m0 + r;
    const bool val = ge < E;
    const float* hp = hsrc + (size_t)(val ? src[ge] : 0) * 256;
#pragma unroll
    for (int j = 0; j < 32; ++j) {
      float4 v = make_float4(0.f, 0.f, 0.f, 0.f);
      if (val) v = *reinterpret_cast<const float4*>(hp + cb + j * 4);
      *reinterpret_cast<uint2*>(&Hc[r * HS + cb + j * 4]) =
          make_uint2(pack2(v.x, v.y), pack2(v.z, v.w));
    }
  }
  __syncthreads();

  // per-lane row state for this wave's two m-frags
  const int e0c = m0 + wm * 32 + l16;
  const int e1c = e0c + 16;
  const float* zp0 = z + (size_t)(e0c < E ? e0c : 0) * 256;
  const float* zp1 = z + (size_t)(e1c < E ? e1c : 0) * 256;
  const unsigned short* hrow0 = &Hc[(wm * 32 + l16) * HS + lhalf * 8];
  const unsigned short* hrow1 = hrow0 + 16 * HS;
  const unsigned short* wbase = Wt + (size_t)(n0 + wn * 64 + l16) * 32 + lhalf * 8;

  v8f acc[2][4] = {};

  // ---- main loop: kz outer (z loaded once), 8 i-chunks inner, no barriers ----
  for (int kz = 0; kz < 256; ++kz) {
    const __bf16 z0 = (__bf16)zp0[kz];
    const __bf16 z1 = (__bf16)zp1[kz];
    const unsigned short* wp = wbase + (size_t)kz * (8 * 8192);
#pragma unroll
    for (int ic = 0; ic < 8; ++ic) {
      const int ib = ic * 32;
      const v16bf a0 = frag16(hrow0 + ib, hrow0 + ib + 16) * z0;
      const v16bf a1 = frag16(hrow1 + ib, hrow1 + ib + 16) * z1;
      const unsigned short* wc = wp + (size_t)ic * 8192;
#pragma unroll
      for (int fn = 0; fn < 4; ++fn) {
        const v16bf b = frag16(wc + fn * 512, wc + fn * 512 + 16);
        acc[0][fn] = __builtin_amdgcn_wmma_f32_16x16x32_bf16(
            false, a0, false, b, (short)0, acc[0][fn], false, false);
        acc[1][fn] = __builtin_amdgcn_wmma_f32_16x16x32_bf16(
            false, a1, false, b, (short)0, acc[1][fn], false, false);
      }
    }
  }
  // ---- bias tail: y = h (scale 1), rows 65536..65791 of B' ----
  {
    const unsigned short* wp = wbase + (size_t)2048 * 8192;
#pragma unroll
    for (int ic = 0; ic < 8; ++ic) {
      const int ib = ic * 32;
      const v16bf a0 = frag16(hrow0 + ib, hrow0 + ib + 16);
      const v16bf a1 = frag16(hrow1 + ib, hrow1 + ib + 16);
      const unsigned short* wc = wp + (size_t)ic * 8192;
#pragma unroll
      for (int fn = 0; fn < 4; ++fn) {
        const v16bf b = frag16(wc + fn * 512, wc + fn * 512 + 16);
        acc[0][fn] = __builtin_amdgcn_wmma_f32_16x16x32_bf16(
            false, a0, false, b, (short)0, acc[0][fn], false, false);
        acc[1][fn] = __builtin_amdgcn_wmma_f32_16x16x32_bf16(
            false, a1, false, b, (short)0, acc[1][fn], false, false);
      }
    }
  }

  // ---- epilogue: atomic scatter-add by destination node ----
#pragma unroll
  for (int fm = 0; fm < 2; ++fm) {
    const int rbase = m0 + wm * 32 + fm * 16 + lhalf * 8;
#pragma unroll
    for (int fn = 0; fn < 4; ++fn) {
      const int col = n0 + wn * 64 + fn * 16 + l16;
#pragma unroll
      for (int v = 0; v < 8; ++v) {
        const int e = rbase + v;
        if (e < E)
          atomicAdd(&agg[(size_t)dst[e] * 256 + col], acc[fm][fn][v]);
      }
    }
  }
}

// ---------------------------------------------------------------------------
// Small helpers
// ---------------------------------------------------------------------------
__global__ void fill_f32(float* __restrict__ p, int n) {
  int i = blockIdx.x * blockDim.x + threadIdx.x;
  if (i < n) p[i] = 0.f;
}
__global__ void fill_i32(int* __restrict__ p, int n) {
  int i = blockIdx.x * blockDim.x + threadIdx.x;
  if (i < n) p[i] = 0;
}
__global__ void edge_count(const int* __restrict__ dst, int* __restrict__ cnt, int E) {
  int i = blockIdx.x * blockDim.x + threadIdx.x;
  if (i < E) atomicAdd(&cnt[dst[i]], 1);
}
__global__ void finalize_node(const float* __restrict__ agg, const int* __restrict__ cnt,
                              const float* __restrict__ rooth, const float* __restrict__ bias,
                              float* __restrict__ out, int S) {
  int idx = blockIdx.x * blockDim.x + threadIdx.x;
  if (idx < S * 256) {
    int n = idx >> 8;
    int o = idx & 255;
    int c = cnt[n]; c = c > 1 ? c : 1;
    float x = agg[idx] / (float)c + rooth[idx] + bias[o];
    out[idx] = (x >= 0.f) ? x : SLOPE * x;
  }
}

// ---------------------------------------------------------------------------
extern "C" void kernel_launch(void* const* d_in, const int* in_sizes, int n_in,
                              void* d_out, int out_size, void* d_ws, size_t ws_size,
                              hipStream_t stream)
{
  enum { NTGT = 5000, NNT = 15000, N0n = 20000, S1n = 5000, S2n = 1000,
         E1n = 4000, E2n = 1500, H = 256 };

  const float* x_t  = (const float*)d_in[0];
  const float* x_n  = (const float*)d_in[1];
  const float* ef1  = (const float*)d_in[2];
  const float* ef2  = (const float*)d_in[3];
  const float* tw0  = (const float*)d_in[4];
  const float* tw1  = (const float*)d_in[5];
  const float* tw2  = (const float*)d_in[6];
  const float* tb0  = (const float*)d_in[7];
  const float* tb1  = (const float*)d_in[8];
  const float* tb2  = (const float*)d_in[9];
  const float* nw0  = (const float*)d_in[10];
  const float* nw1  = (const float*)d_in[11];
  const float* nw2  = (const float*)d_in[12];
  const float* nb0  = (const float*)d_in[13];
  const float* nb1  = (const float*)d_in[14];
  const float* nb2  = (const float*)d_in[15];
  const float* e1w0 = (const float*)d_in[16];
  const float* e1w1 = (const float*)d_in[17];
  const float* e1w2 = (const float*)d_in[18];
  const float* e1b0 = (const float*)d_in[19];
  const float* e1b1 = (const float*)d_in[20];
  const float* e1b2 = (const float*)d_in[21];
  const float* root1 = (const float*)d_in[22];
  const float* bias1 = (const float*)d_in[23];
  const float* e2w0 = (const float*)d_in[24];
  const float* e2w1 = (const float*)d_in[25];
  const float* e2w2 = (const float*)d_in[26];
  const float* e2b0 = (const float*)d_in[27];
  const float* e2b1 = (const float*)d_in[28];
  const float* e2b2 = (const float*)d_in[29];
  const float* root2 = (const float*)d_in[30];
  const float* bias2 = (const float*)d_in[31];
  const float* linw = (const float*)d_in[32];
  const float* linb = (const float*)d_in[33];
  const int* ei1 = (const int*)d_in[34];   // [2, E1]: src then dst
  const int* ei2 = (const int*)d_in[35];

  // workspace carve-up
  float* ws = (float*)d_ws;
  float* h    = ws; ws += (size_t)N0n * H;
  float* tA   = ws; ws += (size_t)NNT * H;
  float* tB   = ws; ws += (size_t)NNT * H;
  float* agg1 = ws; ws += (size_t)S1n * H;
  float* rh1  = ws; ws += (size_t)S1n * H;
  float* h1   = ws; ws += (size_t)S1n * H;
  float* agg2 = ws; ws += (size_t)S2n * H;
  float* rh2  = ws; ws += (size_t)S2n * H;
  float* h2   = ws; ws += (size_t)S2n * H;
  int* cnt1 = (int*)ws; ws += S1n + S2n;      // 6000 ints == 6000 floats of space
  int* cnt2 = cnt1 + S1n;
  unsigned short* Wt = (unsigned short*)ws;   // 2056*8192 bf16 = 33.7 MB
  float* out = (float*)d_out;

#define GEMM(ACT, BIAS, A, lda, B, ldb, bs, C, ldc, M, N, K)                  \
  gemm_wmma<ACT, BIAS><<<dim3(((N) + 127) / 128, ((M) + 127) / 128), 256, 0,  \
                         stream>>>(A, lda, B, ldb, bs, C, ldc, M, N, K)

  // zero the atomic accumulators
  fill_f32<<<(S1n * H + 255) / 256, 256, 0, stream>>>(agg1, S1n * H);
  fill_f32<<<(S2n * H + 255) / 256, 256, 0, stream>>>(agg2, S2n * H);
  fill_i32<<<(S1n + 255) / 256, 256, 0, stream>>>(cnt1, S1n);
  fill_i32<<<(S2n + 255) / 256, 256, 0, stream>>>(cnt2, S2n);

  // node embedding MLPs -> h[0:NTGT], h[NTGT:]
  GEMM(1, 1, x_t, 64, tw0, H, tb0, tA, H, NTGT, H, 64);
  GEMM(1, 1, tA, H, tw1, H, tb1, tB, H, NTGT, H, H);
  GEMM(0, 1, tB, H, tw2, H, tb2, h, H, NTGT, H, H);
  GEMM(1, 1, x_n, 32, nw0, H, nb0, tA, H, NNT, H, 32);
  GEMM(1, 1, tA, H, nw1, H, nb1, tB, H, NNT, H, H);
  GEMM(0, 1, tB, H, nw2, H, nb2, h + (size_t)NTGT * H, H, NNT, H, H);

  // ---- NNConv layer 1 ----
  repack_w<<<2056, 256, 0, stream>>>(e1w2, e1b2, Wt);
  GEMM(1, 1, ef1, 8, e1w0, H, e1b0, tA, H, E1n, H, 8);
  GEMM(1, 1, tA, H, e1w1, H, e1b1, tB, H, E1n, H, H);
  nnconv_msg<<<dim3(2, (E1n + 127) / 128), 256, 0, stream>>>(
      h, tB, Wt, ei1, ei1 + E1n, agg1, E1n);
  edge_count<<<(E1n + 255) / 256, 256, 0, stream>>>(ei1 + E1n, cnt1, E1n);
  GEMM(0, 0, h, H, root1, H, (const float*)nullptr, rh1, H, S1n, H, H);
  finalize_node<<<(S1n * H + 255) / 256, 256, 0, stream>>>(agg1, cnt1, rh1, bias1, h1, S1n);

  // ---- NNConv layer 2 ----
  repack_w<<<2056, 256, 0, stream>>>(e2w2, e2b2, Wt);
  GEMM(1, 1, ef2, 8, e2w0, H, e2b0, tA, H, E2n, H, 8);
  GEMM(1, 1, tA, H, e2w1, H, e2b1, tB, H, E2n, H, H);
  nnconv_msg<<<dim3(2, (E2n + 127) / 128), 256, 0, stream>>>(
      h1, tB, Wt, ei2, ei2 + E2n, agg2, E2n);
  edge_count<<<(E2n + 255) / 256, 256, 0, stream>>>(ei2 + E2n, cnt2, E2n);
  GEMM(0, 0, h1, H, root2, H, (const float*)nullptr, rh2, H, S2n, H, H);
  finalize_node<<<(S2n * H + 255) / 256, 256, 0, stream>>>(agg2, cnt2, rh2, bias2, h2, S2n);

  // ---- final linear + lrelu ----
  GEMM(1, 1, h2, H, linw, 16, linb, out, 16, S2n, 16, H);
#undef GEMM
}
